// GraphTransformerLayer_2594160247141
// MI455X (gfx1250) — matrix operand order repaired and checked by reading, not compiled, not run on hardware
//
#include <hip/hip_runtime.h>
#include <hip/hip_bf16.h>

#define NN 40000
#define EE 320000
#define DDIM 256
#define HH 8
#define RR 9
#define BB 40

typedef __attribute__((ext_vector_type(16))) __bf16 v16bf;
typedef __attribute__((ext_vector_type(8)))  float  v8f;

// ---------- bf16 helpers (raw-bit storage, round-to-nearest-even) ----------
__device__ __forceinline__ unsigned short f2bf(float f) {
    unsigned int u = __float_as_uint(f);
    u += 0x7FFFu + ((u >> 16) & 1u);
    return (unsigned short)(u >> 16);
}

// ---------- GraphNorm: per-graph mean/std over npg nodes, out = bf16 ----------
__global__ __launch_bounds__(256)
void graphnorm_k(const float* __restrict__ x, const float* __restrict__ w,
                 const float* __restrict__ b, const float* __restrict__ ms,
                 unsigned short* __restrict__ out, int npg)
{
    const int g = blockIdx.x, d = threadIdx.x;
    const float* xg = x + (size_t)g * npg * DDIM + d;
    float s1 = 0.f, s2 = 0.f;
    for (int i = 0; i < npg; ++i) { float v = xg[(size_t)i * DDIM]; s1 += v; s2 += v * v; }
    const float mean = s1 / (float)npg;
    const float mm   = mean * ms[d];
    // E[(x-mm)^2] = E[x^2] - 2*mm*E[x] + mm^2
    const float var  = s2 / (float)npg - 2.f * mm * mean + mm * mm;
    const float inv  = rsqrtf(var + 1e-6f);
    const float wd = w[d], bd = b[d];
    unsigned short* og = out + (size_t)g * npg * DDIM + d;
    for (int i = 0; i < npg; ++i)
        og[(size_t)i * DDIM] = f2bf(wd * (xg[(size_t)i * DDIM] - mm) * inv + bd);
}

// ---------- Wt_r[e][d] = sum_b comb[r,b] * bases[b][d][e]  (f32 -> bf16, TRANSPOSED) ----------
__global__ __launch_bounds__(256)
void combine_w_k(const float* __restrict__ bases, const float* __restrict__ comb,
                 unsigned short* __restrict__ Wt)
{
    const long i = (long)blockIdx.x * 256 + threadIdx.x;     // over R*D*D (source order)
    if (i >= (long)RR * DDIM * DDIM) return;
    const int  r = (int)(i / (DDIM * DDIM));
    const long p = i - (long)r * DDIM * DDIM;                // p = d*D + e
    const int  d = (int)(p >> 8), e = (int)(p & 255);
    float acc = 0.f;
    #pragma unroll
    for (int bb = 0; bb < RR; ++bb)
        acc += comb[r * RR + bb] * bases[(long)bb * DDIM * DDIM + p];
    Wt[(size_t)r * DDIM * DDIM + (size_t)e * DDIM + d] = f2bf(acc);
}

// ---------- f32 (rows x cols) -> bf16 transposed (cols x rows) ----------
__global__ __launch_bounds__(256)
void cvt_bf16_t_k(const float* __restrict__ in, unsigned short* __restrict__ out,
                  int rows, int cols)
{
    const long i = (long)blockIdx.x * 256 + threadIdx.x;
    if (i >= (long)rows * cols) return;
    const int r = (int)(i / cols), c = (int)(i % cols);
    out[(size_t)c * rows + r] = f2bf(in[i]);
}

// ---------- WMMA GEMM: C(MxN) = A(MxK,bf16) @ Bt(NxK,bf16 transposed) (+bias,relu) ----------
// One wave32 owns a 16x32 output tile (two 16x16 WMMA tiles sharing the A fragment).
// Fully unrolled K loop (KK template param); all fragment loads are b128.
// D layout per CDNA5 ISA 7.12.2; A/B use contiguous k = 16*lh + s.
template<int KK>
__global__ __launch_bounds__(256)
void gemm_wmma_k(const unsigned short* __restrict__ A, const unsigned short* __restrict__ Bt,
                 float* __restrict__ Cf, unsigned short* __restrict__ Cb,
                 const float* __restrict__ bias, int M, int Nd, int relu)
{
    const int wave = threadIdx.x >> 5, lane = threadIdx.x & 31;
    const int tilesN = Nd >> 5;                          // 32-wide column tiles
    const long tileId = (long)blockIdx.x * 8 + wave;
    if (tileId >= (long)(M >> 4) * tilesN) return;
    const int tm = (int)(tileId / tilesN), tn = (int)(tileId % tilesN);
    const int lh = lane >> 4, ln = lane & 15;

    const uint4* Arow  = (const uint4*)(A  + (size_t)(tm * 16 + ln) * KK);
    const int col0 = tn * 32 + ln;
    const uint4* Brow0 = (const uint4*)(Bt + (size_t)col0 * KK);
    const uint4* Brow1 = (const uint4*)(Bt + (size_t)(col0 + 16) * KK);
    __builtin_prefetch(Brow0, 0, 1);
    __builtin_prefetch(Brow1, 0, 1);

    v8f c0 = {0.f, 0.f, 0.f, 0.f, 0.f, 0.f, 0.f, 0.f};
    v8f c1 = {0.f, 0.f, 0.f, 0.f, 0.f, 0.f, 0.f, 0.f};
    #pragma unroll
    for (int k0 = 0; k0 < KK; k0 += 32) {
        union { v16bf v; uint4 q[2]; } a, b0, b1;
        const int q0 = k0 >> 3;                          // uint4 index at this k-step
        // A frag: k = k0 + lh*16 + 2j -> 16 contiguous bf16
        a.q[0]  = Arow[q0 + (lh << 1)];
        a.q[1]  = Arow[q0 + (lh << 1) + 1];
        // B frag (transposed rows): k = k0 + lh*16 + 2j -> 16 contiguous bf16
        b0.q[0] = Brow0[q0 + (lh << 1)];
        b0.q[1] = Brow0[q0 + (lh << 1) + 1];
        b1.q[0] = Brow1[q0 + (lh << 1)];
        b1.q[1] = Brow1[q0 + (lh << 1) + 1];
        c0 = __builtin_amdgcn_wmma_f32_16x16x32_bf16(false, a.v, false, b0.v,
                                                     (short)0, c0, false, false);
        c1 = __builtin_amdgcn_wmma_f32_16x16x32_bf16(false, a.v, false, b1.v,
                                                     (short)0, c1, false, false);
    }
    const int col1 = col0 + 16;
    const float bv0 = bias ? bias[col0] : 0.f;
    const float bv1 = bias ? bias[col1] : 0.f;
    #pragma unroll
    for (int j = 0; j < 8; ++j) {
        const int m = tm * 16 + (lh << 3) + j;           // D layout: m = vgpr + 8*(lane>=16)
        float v0 = c0[j] + bv0, v1 = c1[j] + bv1;
        if (relu) { v0 = fmaxf(v0, 0.f); v1 = fmaxf(v1, 0.f); }
        if (Cf) { Cf[(size_t)m * Nd + col0] = v0; Cf[(size_t)m * Nd + col1] = v1; }
        else    { Cb[(size_t)m * Nd + col0] = f2bf(v0); Cb[(size_t)m * Nd + col1] = f2bf(v1); }
    }
}

// ---------- per-relation segment-sum scatter: out[dst] += hW[src] where etype==r ----------
__global__ __launch_bounds__(256)
void scatter_rel_k(const float* __restrict__ hW, const int* __restrict__ src,
                   const int* __restrict__ dst, const int* __restrict__ et,
                   int r, float* __restrict__ out)
{
    const long t = (long)blockIdx.x * 256 + threadIdx.x;
    const int  e = (int)(t >> 5);
    if (e >= EE) return;
    if (et[e] != r) return;
    const int lane = (int)(t & 31);
    const float* srow = hW  + (size_t)src[e] * DDIM;
    float*       drow = out + (size_t)dst[e] * DDIM;
    #pragma unroll
    for (int i = 0; i < 8; ++i) {
        const int d = lane + i * 32;
        atomicAdd(&drow[d], srow[d]);
    }
}

// ---------- out = relu(acc + bias[d]) ----------
__global__ __launch_bounds__(256)
void bias_relu_k(float* __restrict__ x, const float* __restrict__ bias, long n)
{
    const long i = (long)blockIdx.x * 256 + threadIdx.x;
    if (i >= n) return;
    x[i] = fmaxf(x[i] + bias[i & (DDIM - 1)], 0.f);
}

// ---------- attention edge pass: one wave per edge; per-head dot via wave32 shuffle ----------
__global__ __launch_bounds__(256)
void attn_edge_k(const float* __restrict__ Q, const float* __restrict__ K,
                 const float* __restrict__ V, const int* __restrict__ src,
                 const int* __restrict__ dst, float* __restrict__ wV,
                 float* __restrict__ z)
{
    const long t = (long)blockIdx.x * 256 + threadIdx.x;
    const int  e = (int)(t >> 5);
    if (e >= EE) return;
    const int lane = (int)(t & 31);
    const int s = src[e], d0 = dst[e];
    const float* Ks  = K  + (size_t)s  * DDIM;
    const float* Qd  = Q  + (size_t)d0 * DDIM;
    const float* Vs  = V  + (size_t)s  * DDIM;
    float*       wVd = wV + (size_t)d0 * DDIM;
    const float scale = 0.17677669529663687f;   // 1/sqrt(32)
    #pragma unroll
    for (int hh = 0; hh < HH; ++hh) {
        const int idx = hh * 32 + lane;
        float p = Ks[idx] * Qd[idx];
        #pragma unroll
        for (int off = 16; off > 0; off >>= 1) p += __shfl_xor(p, off, 32);
        const float sc = __expf(fminf(fmaxf(p * scale, -10.f), 10.f));
        if (lane == 0) atomicAdd(&z[(size_t)d0 * HH + hh], sc);
        atomicAdd(&wVd[idx], sc * Vs[idx]);
    }
}

// ---------- attn = wV / (z + 1e-6) -> bf16 ----------
__global__ __launch_bounds__(256)
void attn_fin_k(const float* __restrict__ wV, const float* __restrict__ z,
                unsigned short* __restrict__ out)
{
    const long i = (long)blockIdx.x * 256 + threadIdx.x;
    if (i >= (long)NN * DDIM) return;
    const int node = (int)(i >> 8), d = (int)(i & 255);
    out[i] = f2bf(wV[i] / (z[(size_t)node * HH + (d >> 5)] + 1e-6f));
}

extern "C" void kernel_launch(void* const* d_in, const int* in_sizes, int n_in,
                              void* d_out, int out_size, void* d_ws, size_t ws_size,
                              hipStream_t stream)
{
    (void)in_sizes; (void)n_in; (void)out_size; (void)ws_size;
    const float* h     = (const float*)d_in[0];
    const int*   src   = (const int*)d_in[1];
    const int*   dst   = (const int*)d_in[2];
    const int*   etype = (const int*)d_in[3];
    const float* bases_[3] = { (const float*)d_in[4], (const float*)d_in[7],  (const float*)d_in[10] };
    const float* combs_[3] = { (const float*)d_in[5], (const float*)d_in[8],  (const float*)d_in[11] };
    const float* biass_[3] = { (const float*)d_in[6], (const float*)d_in[9],  (const float*)d_in[12] };
    const float* gn1_w = (const float*)d_in[13];
    const float* gn1_b = (const float*)d_in[14];
    const float* gn1_ms= (const float*)d_in[15];
    const float* gn2_w = (const float*)d_in[16];
    const float* gn2_b = (const float*)d_in[17];
    const float* gn2_ms= (const float*)d_in[18];
    const float* O_w   = (const float*)d_in[19];
    const float* O_b   = (const float*)d_in[20];
    const float* f1_w  = (const float*)d_in[21];
    const float* f1_b  = (const float*)d_in[22];
    const float* f2_w  = (const float*)d_in[23];
    const float* f2_b  = (const float*)d_in[24];

    char* ws = (char*)d_ws;
    size_t off = 0;
    auto take = [&](size_t bytes) -> char* {
        char* p = ws + off; off += (bytes + 255) & ~(size_t)255; return p;
    };
    unsigned short* hn      = (unsigned short*)take((size_t)NN * DDIM * 2);        // hn, later h2 (bf16)
    unsigned short* Wbuf    = (unsigned short*)take((size_t)RR * DDIM * DDIM * 2); // transposed weights (bf16)
    float*          scratch = (float*)take((size_t)NN * DDIM * 4);                 // hW -> h1 -> FFN mid(bf16)
    float*          Qb      = (float*)take((size_t)NN * DDIM * 4);
    float*          Kb      = (float*)take((size_t)NN * DDIM * 4);
    float*          Vb      = (float*)take((size_t)NN * DDIM * 4);
    float*          wV      = (float*)take((size_t)NN * DDIM * 4);
    float*          z       = (float*)take((size_t)NN * HH * 4);
    unsigned short* attn_bf = (unsigned short*)take((size_t)NN * DDIM * 2);

    hipMemsetAsync(Qb, 0, (size_t)NN * DDIM * 4, stream);
    hipMemsetAsync(Kb, 0, (size_t)NN * DDIM * 4, stream);
    hipMemsetAsync(Vb, 0, (size_t)NN * DDIM * 4, stream);
    hipMemsetAsync(wV, 0, (size_t)NN * DDIM * 4, stream);
    hipMemsetAsync(z,  0, (size_t)NN * HH   * 4, stream);

    // GraphNorm 1 -> bf16
    graphnorm_k<<<BB, 256, 0, stream>>>(h, gn1_w, gn1_b, gn1_ms, hn, NN / BB);

    const int gemmBlkNxD  = ((NN / 16) * (DDIM / 32) + 7) / 8;          // 2500
    const int gemmBlkNx2D = ((NN / 16) * ((2 * DDIM) / 32) + 7) / 8;    // 5000
    const int edgeBlocks  = (EE * 32 + 255) / 256;                      // 40000

    // Relational convs Q/K/V: per relation GEMM into scratch, scatter-add per edge
    float* outs[3] = { Qb, Kb, Vb };
    for (int cI = 0; cI < 3; ++cI) {
        combine_w_k<<<(RR * DDIM * DDIM + 255) / 256, 256, 0, stream>>>(bases_[cI], combs_[cI], Wbuf);
        for (int r = 0; r < RR; ++r) {
            gemm_wmma_k<DDIM><<<gemmBlkNxD, 256, 0, stream>>>(
                hn, Wbuf + (size_t)r * DDIM * DDIM, scratch, nullptr, nullptr,
                NN, DDIM, 0);
            scatter_rel_k<<<edgeBlocks, 256, 0, stream>>>(scratch, src, dst, etype, r, outs[cI]);
        }
        bias_relu_k<<<(int)(((long)NN * DDIM + 255) / 256), 256, 0, stream>>>(
            outs[cI], biass_[cI], (long)NN * DDIM);
    }

    // Edge attention + softmax-normalize
    attn_edge_k<<<edgeBlocks, 256, 0, stream>>>(Qb, Kb, Vb, src, dst, wV, z);
    attn_fin_k<<<(int)(((long)NN * DDIM + 255) / 256), 256, 0, stream>>>(wV, z, attn_bf);

    // O projection -> h1 (f32, scratch)
    cvt_bf16_t_k<<<(int)(((long)DDIM * DDIM + 255) / 256), 256, 0, stream>>>(O_w, Wbuf, DDIM, DDIM);
    gemm_wmma_k<DDIM><<<gemmBlkNxD, 256, 0, stream>>>(attn_bf, Wbuf, scratch, nullptr, O_b,
                                                      NN, DDIM, 0);

    // GraphNorm 2 -> h2 (bf16, reuses hn buffer)
    graphnorm_k<<<BB, 256, 0, stream>>>(scratch, gn2_w, gn2_b, gn2_ms, hn, NN / BB);

    // FFN1: relu(h2 @ W1 + b1) -> bf16 mid (reuses scratch bytes: N*512*2 == N*256*4)
    cvt_bf16_t_k<<<(int)(((long)DDIM * 2 * DDIM + 255) / 256), 256, 0, stream>>>(f1_w, Wbuf, DDIM, 2 * DDIM);
    gemm_wmma_k<DDIM><<<gemmBlkNx2D, 256, 0, stream>>>(hn, Wbuf, nullptr, (unsigned short*)scratch,
                                                       f1_b, NN, 2 * DDIM, 1);

    // FFN2: mid @ W2 + b2 -> d_out (f32)
    cvt_bf16_t_k<<<(int)(((long)2 * DDIM * DDIM + 255) / 256), 256, 0, stream>>>(f2_w, Wbuf, 2 * DDIM, DDIM);
    gemm_wmma_k<2 * DDIM><<<gemmBlkNxD, 256, 0, stream>>>((const unsigned short*)scratch, Wbuf,
                                                          (float*)d_out, nullptr, f2_b,
                                                          NN, DDIM, 0);
}
